// ResidualAttentionBlock_51934744543518
// MI455X (gfx1250) — matrix operand order repaired and checked by reading, not compiled, not run on hardware
//
#include <hip/hip_runtime.h>
#include <hip/hip_bf16.h>

typedef __attribute__((ext_vector_type(16))) _Float16     v16h;
typedef __attribute__((ext_vector_type(8)))  float        v8f;
typedef __attribute__((ext_vector_type(4)))  unsigned int u32x4;
typedef __attribute__((ext_vector_type(8)))  int          i32x8;
typedef __attribute__((ext_vector_type(4)))  int          i32x4;

#define BB    8
#define TT    1000
#define NHH   8
#define DD    64
#define NQ    (NHH*DD)          // 512
#define TOKS  (BB*TT)           // 8000
#define T_TILE 64
#define S_TILE 32
#define NCHUNK ((TT + S_TILE - 1) / S_TILE)   // 32
#define SCALE  0.125f           // 1/sqrt(64)

#if defined(__HIP_DEVICE_COMPILE__) && __has_builtin(__builtin_amdgcn_tensor_load_to_lds) && __has_builtin(__builtin_amdgcn_s_wait_tensorcnt)
#define USE_TDM 1
#else
#define USE_TDM 0
#endif

// 16-bit A-matrix 16x32 element -> K index (ISA 7.12.2):
// VGPR j in 0..3 : K = 8*half + 2*j + e ; VGPR j in 4..7 : K = 16 + 8*half + 2*(j-4) + e
__device__ __forceinline__ int a16_k(int i, int h) {
    int vj = i >> 1, e = i & 1;
    return (vj < 4) ? (h * 8 + vj * 2 + e) : (16 + h * 8 + (vj - 4) * 2 + e);
}

#if USE_TDM
// Issue a 2D TDM tile load (global -> LDS) per ISA 8.3/8.4.
//   dsize_code: 0=1B 1=2B 2=4B 3=8B ; dims/strides in data_size units
//   pad_i code: LDS padding interval 2^(code+1) DWORDs ; pad_a code: (code+1) DWORDs
__device__ __forceinline__ void tdm_load_2d(unsigned lds_off, const void* gptr,
                                            int dsize_code, int dim0, int dim1,
                                            int tile0, int tile1, long stride0,
                                            int pad_i, int pad_a) {
    unsigned long long ga = (unsigned long long)(uintptr_t)gptr;
    u32x4 g0;
    g0[0] = 1u;                                          // count=1 (valid user D#)
    g0[1] = lds_off;                                     // lds_addr (bytes)
    g0[2] = (unsigned)(ga & 0xFFFFFFFFu);                // global_addr[31:0]
    g0[3] = (unsigned)((ga >> 32) & 0x01FFFFFFu) | (2u << 30); // addr[56:32] | type=2
    unsigned long long s0 = (unsigned long long)stride0;
    i32x8 g1;
    g1[0] = (int)(((unsigned)dsize_code << 16) | (1u << 20) |       // data_size | pad_en
                  ((unsigned)pad_i << 22) | ((unsigned)pad_a << 25));
    g1[1] = (int)(((unsigned)dim0 & 0xFFFFu) << 16);                // tensor_dim0[15:0]
    g1[2] = (int)((((unsigned)dim0 >> 16) & 0xFFFFu) |
                  (((unsigned)dim1 & 0xFFFFu) << 16));              // dim0 hi | dim1 lo
    g1[3] = (int)((((unsigned)dim1 >> 16) & 0xFFFFu) |
                  (((unsigned)tile0 & 0xFFFFu) << 16));             // dim1 hi | tile_dim0
    g1[4] = (int)((unsigned)tile1 & 0xFFFFu);                       // tile_dim1, tile_dim2=0
    g1[5] = (int)(s0 & 0xFFFFFFFFu);                                // dim0_stride[31:0]
    g1[6] = (int)((s0 >> 32) & 0xFFFFu);                            // dim0_stride[47:32]
    g1[7] = 0;                                                      // dim1_stride (unused 2D)
    i32x4 z4 = {0, 0, 0, 0};
    i32x8 z8 = {0, 0, 0, 0, 0, 0, 0, 0};
    __builtin_amdgcn_tensor_load_to_lds(g0, g1, z4, z4, z8, 0);
}
#endif

// ---------------------------------------------------------------- zero
__global__ void zero_f32(float* p, int n) {
    int i = blockIdx.x * blockDim.x + threadIdx.x;
    if (i < n) p[i] = 0.0f;
}

// ---------------------------------------------------------------- q/v projection (WMMA GEMM, M=8000 N=512 K=64)
__global__ void qv_proj(const float* __restrict__ x,
                        const float* __restrict__ wq,
                        const float* __restrict__ wv,
                        float* __restrict__ qbuf,
                        _Float16* __restrict__ vbuf) {
    const int tid  = threadIdx.x;
    const int lane = tid & 31;
    const int wid  = tid >> 5;
    int tile = blockIdx.x * 8 + wid;          // 0 .. 31999
    const int which = tile >= (500 * 32);     // 0 = q, 1 = v
    if (which) tile -= 500 * 32;
    const int mt = tile >> 5;                 // 0..499 (16-token tiles)
    const int nt = tile & 31;                 // 0..31  (16-output tiles)
    const float* W = which ? wv : wq;

    const int col  = lane & 15;
    const int half = lane >> 4;
    const int rowA = mt * 16 + col;           // token index for A rows (m = lane&15)
    const int rowW = nt * 16 + col;           // output row of W for B columns

    v8f c = {};
#pragma unroll
    for (int ks = 0; ks < 2; ++ks) {
        v16h a, b;
#pragma unroll
        for (int i = 0; i < 16; ++i) {
            int ka = ks * 32 + 16 * half + i;
            a[i] = (_Float16)x[rowA * DD + ka];
            int kb = ks * 32 + 16 * half + i;
            b[i] = (_Float16)W[rowW * (DD + 1) + kb];
        }
        c = __builtin_amdgcn_wmma_f32_16x16x32_f16(false, a, false, b,
                                                   (short)0, c, false, false);
    }
    const float bias = W[rowW * (DD + 1) + DD];
    const int tokBase = mt * 16 + 8 * half;   // D row m = r + 8*half
    const int n = nt * 16 + col;
    if (!which) {
#pragma unroll
        for (int r = 0; r < 8; ++r)
            qbuf[(tokBase + r) * NQ + n] = c[r] + bias;
    } else {
#pragma unroll
        for (int r = 0; r < 8; ++r)
            vbuf[(tokBase + r) * NQ + n] = (_Float16)(c[r] + bias);
    }
}

// ---------------------------------------------------------------- fused L1-attention (flash style, per b,h,t-tile)
__global__ void attn(const float* __restrict__ x,
                     const float* __restrict__ wk,
                     const float* __restrict__ qbuf,
                     const _Float16* __restrict__ vbuf,
                     float* __restrict__ acc) {
    __shared__ float    qTile[T_TILE][68];        // 68-float rows: conflict-free + TDM pad
    __shared__ float    kTile[S_TILE][68];
    __shared__ _Float16 pTile[T_TILE][40];
    __shared__ _Float16 vTile[2][S_TILE][68];     // double buffered (TDM overlap)
    __shared__ float    den[T_TILE];

    const int bid = blockIdx.x;               // b*128 + h*16 + tt
    const int b  = bid >> 7;
    const int h  = (bid >> 4) & 7;
    const int tt = bid & 15;
    const int t0 = tt * T_TILE;
    const int tid  = threadIdx.x;
    const int lane = tid & 31;
    const int wid  = tid >> 5;
    const long bBase = (long)b * TT;
    const float* wkh = wk + h * DD;

#if USE_TDM
    // TDM: q tile (64x64 f32, row stride 512, pad 64 DW rows by +4 DW -> stride 68)
    //      v chunk 0 (32x64 f16, row stride 512, pad 32 DW rows by +2 DW -> stride 68 halfs)
    // OOB rows (token >= TT) are zero-filled by the TDM.
    if (wid == 0) {
        tdm_load_2d((unsigned)(uintptr_t)&qTile[0][0],
                    qbuf + (bBase + t0) * NQ + h * DD,
                    /*dsize*/2, DD, TT - t0, DD, T_TILE, NQ, /*pad*/5, 3);
        tdm_load_2d((unsigned)(uintptr_t)&vTile[0][0][0],
                    vbuf + bBase * NQ + h * DD,
                    /*dsize*/1, DD, TT, DD, S_TILE, NQ, /*pad*/4, 1);
        __builtin_amdgcn_s_wait_tensorcnt(0);
    }
#else
    for (int i = tid; i < T_TILE * DD; i += 256) {
        int t = i >> 6, w = i & 63;
        int gt = t0 + t; if (gt > TT - 1) gt = TT - 1;
        qTile[t][w] = qbuf[(bBase + gt) * NQ + h * DD + w];
    }
#endif
    if (tid < T_TILE) den[tid] = 0.0f;

    // WMMA tile assignment: wave -> A row-tile mt, two column tiles nt0/nt1
    const int mt  = wid >> 1;
    const int nt0 = (wid & 1) * 2;
    const int nt1 = nt0 + 1;
    const int col  = lane & 15;
    const int half = lane >> 4;
    v8f c0 = {}, c1 = {};

    __syncthreads();

    int cur = 0;
    for (int ci = 0; ci < NCHUNK; ++ci) {
        const int s0 = ci * S_TILE;
#if USE_TDM
        // prefetch next v chunk into the other buffer; overlaps score math below
        if (wid == 0 && ci + 1 < NCHUNK) {
            const int s1 = s0 + S_TILE;
            tdm_load_2d((unsigned)(uintptr_t)&vTile[1 - cur][0][0],
                        vbuf + (bBase + s1) * NQ + h * DD,
                        /*dsize*/1, DD, TT - s1, DD, S_TILE, NQ, /*pad*/4, 1);
        }
#endif
        // stage k (= x*wk); invalid sources -> k huge so exp -> 0
        for (int i = tid; i < S_TILE * DD; i += 256) {
            int s = i >> 6, w = i & 63;
            int sIdx = s0 + s;
            float kv = 1e30f;
            if (sIdx < TT) kv = x[(bBase + sIdx) * DD + w] * wkh[w];
            kTile[s][w] = kv;
#if !USE_TDM
            _Float16 vv = (_Float16)0.0f;
            if (sIdx < TT) vv = vbuf[(bBase + sIdx) * NQ + h * DD + w];
            vTile[cur][s][w] = vv;
#endif
        }
        __syncthreads();

        // L1 scores: each thread one source s, 8 dest tokens
        {
            const int s  = tid & 31;
            const int tg = tid >> 5;
            float sum[8] = {0, 0, 0, 0, 0, 0, 0, 0};
            for (int w = 0; w < DD; w += 4) {
                float4 kk = *(const float4*)&kTile[s][w];
#pragma unroll
                for (int j = 0; j < 8; ++j) {
                    float4 qq = *(const float4*)&qTile[tg * 8 + j][w];
                    sum[j] += fabsf(qq.x - kk.x) + fabsf(qq.y - kk.y) +
                              fabsf(qq.z - kk.z) + fabsf(qq.w - kk.w);
                }
            }
#pragma unroll
            for (int j = 0; j < 8; ++j) {
                float p = __expf(-SCALE * sum[j]);
                pTile[tg * 8 + j][s] = (_Float16)p;
            }
        }
        __syncthreads();

        // streaming denominator (scores <= 0, so no max tracking needed)
        if (tid < T_TILE) {
            float d = 0.0f;
#pragma unroll
            for (int s = 0; s < S_TILE; ++s) d += (float)pTile[tid][s];
            den[tid] += d;
        }

        // P(16x32) x V(32x16) per wave, two N tiles
        {
            v16h a, bA, bB;
#pragma unroll
            for (int i = 0; i < 16; ++i) {
                a[i]  = pTile[mt * 16 + col][16 * half + i];
                bA[i] = vTile[cur][16 * half + i][nt0 * 16 + col];
                bB[i] = vTile[cur][16 * half + i][nt1 * 16 + col];
            }
            c0 = __builtin_amdgcn_wmma_f32_16x16x32_f16(false, a, false, bA,
                                                        (short)0, c0, false, false);
            c1 = __builtin_amdgcn_wmma_f32_16x16x32_f16(false, a, false, bB,
                                                        (short)0, c1, false, false);
        }
#if USE_TDM
        if (wid == 0) __builtin_amdgcn_s_wait_tensorcnt(0);   // next v chunk landed
#endif
        __syncthreads();
        cur ^= 1;
    }

    // normalize by (1 + den) and accumulate over heads
#pragma unroll
    for (int r = 0; r < 8; ++r) {
        int t  = mt * 16 + 8 * half + r;
        int gt = t0 + t;
        if (gt < TT) {
            float inv = 1.0f / (1.0f + den[t]);
            atomicAdd(&acc[(bBase + gt) * DD + nt0 * 16 + col], c0[r] * inv);
            atomicAdd(&acc[(bBase + gt) * DD + nt1 * 16 + col], c1[r] * inv);
        }
    }
}

// ---------------------------------------------------------------- epilogue: relu -> wf linear -> residual
__global__ void epilogue(const float* __restrict__ x,
                         const float* __restrict__ wf,
                         const float* __restrict__ acc,
                         float* __restrict__ out) {
    __shared__ float r[4][DD];
    const int tid = threadIdx.x;
    const int tl  = tid >> 6;
    const int w   = tid & 63;
    const int tok = blockIdx.x * 4 + tl;      // 2000*4 == 8000 exactly
    r[tl][w] = fmaxf(acc[tok * DD + w], 0.0f);
    __syncthreads();
    float sum = wf[w * (DD + 1) + DD];        // bias
#pragma unroll 8
    for (int i = 0; i < DD; ++i) sum += wf[w * (DD + 1) + i] * r[tl][i];
    out[tok * DD + w] = x[tok * DD + w] + sum;
}

// ---------------------------------------------------------------- launch
extern "C" void kernel_launch(void* const* d_in, const int* in_sizes, int n_in,
                              void* d_out, int out_size, void* d_ws, size_t ws_size,
                              hipStream_t stream) {
    const float* x  = (const float*)d_in[0];
    const float* wq = (const float*)d_in[1];
    const float* wv = (const float*)d_in[2];
    const float* wk = (const float*)d_in[3];
    const float* wf = (const float*)d_in[4];
    float* out = (float*)d_out;

    char* ws = (char*)d_ws;
    float*    qbuf = (float*)ws;                                   // 8000*512 f32 = 16,384,000 B
    _Float16* vbuf = (_Float16*)(ws + (size_t)TOKS * NQ * 4);      //  8,192,000 B
    float*    acc  = (float*)(ws + (size_t)TOKS * NQ * 6);         //  2,048,000 B

    const int accN = TOKS * DD;                                    // 512000
    zero_f32<<<(accN + 255) / 256, 256, 0, stream>>>(acc, accN);
    qv_proj<<<(2 * 500 * 32) / 8, 256, 0, stream>>>(x, wq, wv, qbuf, vbuf);
    attn<<<BB * NHH * (TT + T_TILE - 1) / T_TILE, 256, 0, stream>>>(x, wk, qbuf, vbuf, acc);
    epilogue<<<TOKS / 4, 256, 0, stream>>>(x, wf, acc, out);
}
